// GCN_37718402794123
// MI455X (gfx1250) — matrix operand order repaired and checked by reading, not compiled, not run on hardware
//
#include <hip/hip_runtime.h>
#include <math.h>

// ---------------------------------------------------------------------------
// GCN forward for MI455X (gfx1250, wave32, WMMA).
// Layers 2-4 GEMMs use v_wmma_f32_16x16x32_bf16; bias+self-loop fused into the
// GEMM epilogue; weight staging uses async global->LDS when available.
// ---------------------------------------------------------------------------

typedef __attribute__((ext_vector_type(16))) __bf16 v16bf;
typedef __attribute__((ext_vector_type(8)))  float  v8f;
typedef int v4i __attribute__((ext_vector_type(4)));

#define HDIM 128
#define GDIM 256
#define CDIM 5
#define LDSW_STRIDE 136   // 128 + 8 pad: avoids LDS bank conflicts on column reads

#if defined(__gfx1250__) && __has_builtin(__builtin_amdgcn_global_load_async_to_lds_b128) && __has_builtin(__builtin_amdgcn_s_wait_asynccnt)
#define GCN_HAVE_ASYNC_LDS 1
#else
#define GCN_HAVE_ASYNC_LDS 0
#endif

// --------------------------- degree / normalization ------------------------

__global__ void gcn_deg_init(float* __restrict__ deg, int n) {
    int i = blockIdx.x * 256 + threadIdx.x;
    if (i < n) deg[i] = 1.0f;               // self-loop contributes 1
}

__global__ void gcn_deg_accum(const int* __restrict__ dst, float* __restrict__ deg, int ne) {
    int e = blockIdx.x * 256 + threadIdx.x;
    if (e < ne) unsafeAtomicAdd(&deg[dst[e]], 1.0f);
}

__global__ void gcn_dinv(const float* __restrict__ deg, float* __restrict__ dinv, int n) {
    int i = blockIdx.x * 256 + threadIdx.x;
    if (i < n) dinv[i] = rsqrtf(deg[i]);
}

// ------------------ layer-1 GEMM (K=4), fused bias+self-loop ---------------

__global__ __launch_bounds__(256)
void gcn_gemm_l1(const float* __restrict__ x, const float* __restrict__ W1,
                 const float* __restrict__ b1, const float* __restrict__ dinv,
                 float* __restrict__ T, float* __restrict__ Hout, int n_nodes) {
    __shared__ float w[4 * HDIM + HDIM];            // W1 (4x128) then b1 (128)
    for (int i = threadIdx.x; i < 5 * HDIM; i += 256)
        w[i] = (i < 4 * HDIM) ? W1[i] : b1[i - 4 * HDIM];
    __syncthreads();
    size_t idx = (size_t)blockIdx.x * 256 + threadIdx.x;
    size_t total = (size_t)n_nodes * HDIM;
    if (idx >= total) return;
    int nrow = (int)(idx >> 7);
    int c    = (int)(idx & 127);
    float4 xf = *(const float4*)(x + (size_t)nrow * 4);
    float t = xf.x * w[c] + xf.y * w[HDIM + c] + xf.z * w[2 * HDIM + c] + xf.w * w[3 * HDIM + c];
    T[idx] = t;
    float di = dinv[nrow];
    Hout[idx] = w[4 * HDIM + c] + di * di * t;      // bias + self-loop message
}

// ------ WMMA GEMM: T = relu?(Hin) @ W ; Hout = b + dinv^2*T  (fused) -------
// Block = 256 threads (8 waves). Block computes 128 rows x 128 cols.
// Wave w handles rows [blk*128 + 16w, +16), all 8 column tiles, K=128.

template <bool RELU_IN>
__global__ __launch_bounds__(256)
void gcn_gemm_wmma(const float* __restrict__ Hin, const float* __restrict__ W,
                   const float* __restrict__ b, const float* __restrict__ dinv,
                   float* __restrict__ T, float* __restrict__ Hout, int n_nodes) {
    __shared__ __bf16 wt[HDIM * LDSW_STRIDE];   // W transposed: wt[n][k], bf16

#if GCN_HAVE_ASYNC_LDS
    // Stage fp32 W into LDS with the gfx1250 async global->LDS path, then
    // convert/transpose LDS->LDS.
    __shared__ float wraw[HDIM * HDIM];
    {
        const float* gp = W + threadIdx.x * 4;      // 16 B per async op
        float*       lp = wraw + threadIdx.x * 4;
        #pragma unroll
        for (int i = 0; i < 16; ++i) {
            __builtin_amdgcn_global_load_async_to_lds_b128(
                (__attribute__((address_space(1))) v4i*)(gp + i * 1024),
                (__attribute__((address_space(3))) v4i*)(lp + i * 1024),
                0, 0);
        }
        __builtin_amdgcn_s_wait_asynccnt(0);
    }
    __syncthreads();
    for (int i = threadIdx.x; i < HDIM * HDIM; i += 256) {
        int k = i >> 7, n = i & 127;
        wt[n * LDSW_STRIDE + k] = (__bf16)wraw[i];
    }
#else
    for (int i = threadIdx.x; i < HDIM * HDIM; i += 256) {
        int k = i >> 7, n = i & 127;
        wt[n * LDSW_STRIDE + k] = (__bf16)W[i];
    }
#endif
    __syncthreads();

    const int lane   = threadIdx.x & 31;
    const int laneHi = lane >> 4;     // 0/1: which K half-group this lane owns
    const int laneLo = lane & 15;
    const int m0     = blockIdx.x * 128 + (threadIdx.x >> 5) * 16;

    v8f acc[8] = {};

    int arow = m0 + laneLo;
    if (arow >= n_nodes) arow = n_nodes - 1;     // clamp; masked at store
    const float* aptr = Hin + (size_t)arow * HDIM + laneHi * 8;

    #pragma unroll
    for (int kk = 0; kk < 4; ++kk) {
        // A fragment (16x32 bf16): lane row = laneLo, kbase = 8*laneHi.
        // elems 0..7  -> K = 32*kk + kbase + j
        // elems 8..15 -> K = 32*kk + kbase + 16 + (j-8)
        v16bf afrag;
        const float* p0 = aptr + kk * 32;
        #pragma unroll
        for (int j = 0; j < 8; ++j) {
            float f0 = p0[j];
            float f1 = p0[16 + j];
            if (RELU_IN) { f0 = fmaxf(f0, 0.0f); f1 = fmaxf(f1, 0.0f); }
            afrag[j]     = (__bf16)f0;
            afrag[j + 8] = (__bf16)f1;
        }
        // 8 column tiles; B fragment (32x16 bf16): lane col = laneLo,
        // K = 32*kk + 8*laneHi + j (j<8), 32*kk + 8*laneHi + 16 + (j-8) (j>=8), read from transposed LDS.
        #pragma unroll
        for (int n = 0; n < 8; ++n) {
            v16bf bfrag;
            const __bf16* bp = &wt[(n * 16 + laneLo) * LDSW_STRIDE + kk * 32 + laneHi * 8];
            #pragma unroll
            for (int j = 0; j < 16; ++j) bfrag[j] = bp[j < 8 ? j : j + 8];
            acc[n] = __builtin_amdgcn_wmma_f32_16x16x32_bf16(
                false, afrag, false, bfrag, (short)0, acc[n], false, false);
        }
    }

    // C/D layout: VGPR r, lanes 0-15 -> M=r, lanes 16-31 -> M=8+r; N=laneLo.
    // Fused epilogue: T = t ; Hout = b + dinv^2 * t  (self-loop init).
    float dsq[8];
    bool  ok[8];
    #pragma unroll
    for (int r = 0; r < 8; ++r) {
        int row = m0 + r + 8 * laneHi;
        ok[r] = row < n_nodes;
        float di = dinv[ok[r] ? row : 0];
        dsq[r] = di * di;
    }
    #pragma unroll
    for (int n = 0; n < 8; ++n) {
        int col = n * 16 + laneLo;
        float bc = b[col];
        #pragma unroll
        for (int r = 0; r < 8; ++r) {
            if (ok[r]) {
                int row = m0 + r + 8 * laneHi;
                size_t off = (size_t)row * HDIM + col;
                float t = acc[n][r];
                T[off]    = t;
                Hout[off] = bc + dsq[r] * t;
            }
        }
    }
}

// ---------------- edge aggregation: Hout[dst] += coef * T[src] -------------
// One wave per edge iteration; grid-stride with prefetch of the next edge's
// source row (global_prefetch_b8). Each lane handles 4 channels.

__global__ __launch_bounds__(256)
void gcn_edge_agg(const int* __restrict__ src, const int* __restrict__ dst,
                  const float* __restrict__ dinv, const float* __restrict__ T,
                  float* __restrict__ Hout, int ne) {
    const int lane   = threadIdx.x & 31;
    const int stride = gridDim.x * 8;
    for (int e = blockIdx.x * 8 + (threadIdx.x >> 5); e < ne; e += stride) {
        int s = src[e];
        int d = dst[e];
        int en = e + stride;
        if (en < ne) {
            int sn = src[en];
            __builtin_prefetch(T + (size_t)sn * HDIM + lane * 4, 0, 1);
        }
        float c = dinv[s] * dinv[d];
        float4 v = *(const float4*)(T + (size_t)s * HDIM + lane * 4);
        float* o = Hout + (size_t)d * HDIM + lane * 4;
        unsafeAtomicAdd(o + 0, c * v.x);
        unsafeAtomicAdd(o + 1, c * v.y);
        unsafeAtomicAdd(o + 2, c * v.z);
        unsafeAtomicAdd(o + 3, c * v.w);
    }
}

// ------------------------------ pooling + head -----------------------------

__global__ void gcn_pool_zero(float* __restrict__ pooled, float* __restrict__ cnt) {
    int i = blockIdx.x * 256 + threadIdx.x;
    if (i < GDIM * HDIM) pooled[i] = 0.0f;
    if (i < GDIM) cnt[i] = 0.0f;
}

__global__ __launch_bounds__(128)
void gcn_pool_accum(const float* __restrict__ H4, const int* __restrict__ batch,
                    float* __restrict__ pooled, float* __restrict__ cnt, int n_nodes) {
    int nrow = blockIdx.x;
    if (nrow >= n_nodes) return;
    int c = threadIdx.x;                         // 128 channels
    int g = batch[nrow];
    float v = fmaxf(H4[(size_t)nrow * HDIM + c], 0.0f);   // final relu fused here
    unsafeAtomicAdd(&pooled[g * HDIM + c], v);
    if (c == 0) unsafeAtomicAdd(&cnt[g], 1.0f);
}

__global__ void gcn_head(const float* __restrict__ pooled, const float* __restrict__ cnt,
                         const float* __restrict__ Wl, const float* __restrict__ bl,
                         float* __restrict__ out) {
    int idx = blockIdx.x * 256 + threadIdx.x;
    if (idx >= GDIM * CDIM) return;
    int g = idx / CDIM, c = idx % CDIM;
    float inv = 1.0f / fmaxf(cnt[g], 1.0f);
    float s = 0.0f;
    #pragma unroll 4
    for (int h = 0; h < HDIM; ++h) s += pooled[g * HDIM + h] * Wl[h * CDIM + c];
    float z = s * inv + bl[c];
    out[idx] = 1.0f / (1.0f + expf(-z));
}

// ------------------------------- launch ------------------------------------

extern "C" void kernel_launch(void* const* d_in, const int* in_sizes, int n_in,
                              void* d_out, int out_size, void* d_ws, size_t ws_size,
                              hipStream_t stream) {
    const float* x    = (const float*)d_in[0];
    const int*   ei   = (const int*)d_in[1];
    const int*   batch= (const int*)d_in[2];
    const float* W1   = (const float*)d_in[3];
    const float* b1   = (const float*)d_in[4];
    const float* W2   = (const float*)d_in[5];
    const float* b2   = (const float*)d_in[6];
    const float* W3   = (const float*)d_in[7];
    const float* b3   = (const float*)d_in[8];
    const float* W4   = (const float*)d_in[9];
    const float* b4   = (const float*)d_in[10];
    const float* Wl   = (const float*)d_in[11];
    const float* bl   = (const float*)d_in[12];
    float*       out  = (float*)d_out;

    const int N = in_sizes[0] / 4;      // nodes
    const int E = in_sizes[1] / 2;      // edges
    const int* src = ei;
    const int* dst = ei + E;

    float* ws     = (float*)d_ws;
    float* deg    = ws;
    float* dinv   = deg  + (size_t)N;
    float* T      = dinv + (size_t)N;              // GEMM output (pre-bias messages)
    float* bufA   = T    + (size_t)N * HDIM;
    float* bufB   = bufA + (size_t)N * HDIM;
    float* pooled = bufB + (size_t)N * HDIM;
    float* cnt    = pooled + (size_t)GDIM * HDIM;

    const int nThreads = 256;
    const int nBlkN    = (N + nThreads - 1) / nThreads;
    const int nBlkE    = (E + nThreads - 1) / nThreads;
    const int nBlkNH   = (int)(((size_t)N * HDIM + nThreads - 1) / nThreads);
    const int nBlkGemm = (N + 127) / 128;
    int nBlkEdge = 4096;                 // grid-stride: ~50 edges per wave
    if (nBlkEdge * 8 > E) nBlkEdge = (E + 7) / 8;

    // symmetric normalization coefficients
    gcn_deg_init <<<nBlkN, nThreads, 0, stream>>>(deg, N);
    gcn_deg_accum<<<nBlkE, nThreads, 0, stream>>>(dst, deg, E);
    gcn_dinv     <<<nBlkN, nThreads, 0, stream>>>(deg, dinv, N);

    // layer 1: x(N,4) @ W1 -> T, bufA = b1 + dinv^2*T ; then bufA += edge msgs
    gcn_gemm_l1 <<<nBlkNH, nThreads, 0, stream>>>(x, W1, b1, dinv, T, bufA, N);
    gcn_edge_agg<<<nBlkEdge, nThreads, 0, stream>>>(src, dst, dinv, T, bufA, E);

    // layer 2: relu(A) @ W2 -> T, bufB = b2 + dinv^2*T ; bufB += msgs
    gcn_gemm_wmma<true><<<nBlkGemm, nThreads, 0, stream>>>(bufA, W2, b2, dinv, T, bufB, N);
    gcn_edge_agg       <<<nBlkEdge, nThreads, 0, stream>>>(src, dst, dinv, T, bufB, E);

    // layer 3
    gcn_gemm_wmma<true><<<nBlkGemm, nThreads, 0, stream>>>(bufB, W3, b3, dinv, T, bufA, N);
    gcn_edge_agg       <<<nBlkEdge, nThreads, 0, stream>>>(src, dst, dinv, T, bufA, E);

    // layer 4
    gcn_gemm_wmma<true><<<nBlkGemm, nThreads, 0, stream>>>(bufA, W4, b4, dinv, T, bufB, N);
    gcn_edge_agg       <<<nBlkEdge, nThreads, 0, stream>>>(src, dst, dinv, T, bufB, E);

    // mean pool (relu fused) + sigmoid head
    gcn_pool_zero <<<(GDIM * HDIM + 255) / 256, nThreads, 0, stream>>>(pooled, cnt);
    gcn_pool_accum<<<N, 128, 0, stream>>>(bufB, batch, pooled, cnt, N);
    gcn_head      <<<(GDIM * CDIM + 255) / 256, nThreads, 0, stream>>>(pooled, cnt, Wl, bl, out);
}
